// EnhancedTemporalGraphNetwork_12524124635331
// MI455X (gfx1250) — matrix operand order repaired and checked by reading, not compiled, not run on hardware
//
#include <hip/hip_runtime.h>

// ---------------------------------------------------------------------------
// EnhancedTemporalGraphNetwork on MI455X (gfx1250, wave32, WMMA bf16)
// Weights pre-swizzled into WMMA B-fragment order (2x b128 per fragment);
// A fragments are two ds_load_b128 from row-major LDS tiles. Store path is
// branch-uniform: only the single ragged tail block takes guarded stores.
// ---------------------------------------------------------------------------
#define N_NODES 50000
#define N_EDGES 800000
#define HID 256

typedef __attribute__((ext_vector_type(16))) __bf16 v16bf;
typedef __attribute__((ext_vector_type(8)))  float  v8f;

union BFrag {
    v16bf v;
    uint4 q[2];
    unsigned int u[8];
};

__device__ __forceinline__ unsigned short f2bf(float f) {
    unsigned int u = __float_as_uint(f);
    unsigned int r = u + 0x7FFFu + ((u >> 16) & 1u);
    return (unsigned short)(r >> 16);
}

// -------------------------- elementwise helpers ----------------------------
__global__ void k_zero(float* __restrict__ p, size_t n) {
    size_t i = (size_t)blockIdx.x * blockDim.x + threadIdx.x;
    if (i < n) p[i] = 0.0f;
}

__global__ void k_f2bf(const float* __restrict__ s, unsigned short* __restrict__ d, size_t n) {
    size_t i = (size_t)blockIdx.x * blockDim.x + threadIdx.x;
    if (i < n) d[i] = f2bf(s[i]);
}

__global__ void k_i64_to_i32(const long long* __restrict__ s, int* __restrict__ d, int n) {
    int i = blockIdx.x * blockDim.x + threadIdx.x;
    if (i < n) d[i] = (int)s[i];
}

__global__ void k_deg(const int* __restrict__ dst, float* __restrict__ deg, int E) {
    int e = blockIdx.x * blockDim.x + threadIdx.x;
    if (e < E) atomicAdd(&deg[dst[e]], 1.0f);
}

__global__ void k_invdeg(float* __restrict__ deg, int N) {
    int i = blockIdx.x * blockDim.x + threadIdx.x;
    if (i < N) deg[i] = 1.0f / fmaxf(deg[i], 1.0f);
}

// scatter-add: each thread moves 4 features of one edge; agg[dst] += X[src]
__global__ void k_scatter(const float* __restrict__ X,
                          const int* __restrict__ src,
                          const int* __restrict__ dst,
                          float* __restrict__ agg, int E) {
    long long tid = (long long)blockIdx.x * blockDim.x + threadIdx.x;
    if (tid >= (long long)E * 64) return;
    int e = (int)(tid >> 6);
    int f = ((int)tid & 63) << 2;
    size_t s = (size_t)src[e], d = (size_t)dst[e];
    float4 v = *(const float4*)(X + s * HID + f);
    float* out = agg + d * HID + f;
    atomicAdd(out + 0, v.x);
    atomicAdd(out + 1, v.y);
    atomicAdd(out + 2, v.z);
    atomicAdd(out + 3, v.w);
}

// out_bf16 = bf16(agg * inv_deg[row])
__global__ void k_scale_bf16(const float* __restrict__ agg, const float* __restrict__ invdeg,
                             unsigned short* __restrict__ out, size_t n) {
    size_t i = (size_t)blockIdx.x * blockDim.x + threadIdx.x;
    if (i < n) out[i] = f2bf(agg[i] * invdeg[i >> 8]);
}

__global__ void k_add_tanh(float* __restrict__ hn, const float* __restrict__ h,
                           const float* __restrict__ t, size_t n) {
    size_t i = (size_t)blockIdx.x * blockDim.x + threadIdx.x;
    if (i < n) hn[i] = h[i] + tanhf(t[i]);
}

__global__ void k_sigmoid(float* __restrict__ p, size_t n) {
    size_t i = (size_t)blockIdx.x * blockDim.x + threadIdx.x;
    if (i < n) p[i] = 1.0f / (1.0f + expf(-p[i]));
}

__global__ void k_tanh(float* __restrict__ p, size_t n) {
    size_t i = (size_t)blockIdx.x * blockDim.x + threadIdx.x;
    if (i < n) p[i] = tanhf(p[i]);
}

__global__ void k_mul(float* __restrict__ d, const float* __restrict__ a,
                      const float* __restrict__ b, size_t n) {
    size_t i = (size_t)blockIdx.x * blockDim.x + threadIdx.x;
    if (i < n) d[i] = a[i] * b[i];
}

__global__ void k_gate(float* __restrict__ h, const float* __restrict__ z,
                       const float* __restrict__ hn, const float* __restrict__ ht, size_t n) {
    size_t i = (size_t)blockIdx.x * blockDim.x + threadIdx.x;
    if (i < n) h[i] = (1.0f - z[i]) * hn[i] + z[i] * ht[i];
}

__global__ void k_copy(float* __restrict__ d, const float* __restrict__ s, size_t n) {
    size_t i = (size_t)blockIdx.x * blockDim.x + threadIdx.x;
    if (i < n) d[i] = s[i];
}

// -------- weight repack: f32 [K,256] row-major -> bf16 B-fragment order ----
// Output uint index = ((nt*(K/32) + ks)*32 + lane)*8 + i, holding the packed
// pair W[k][n] | W[k+1][n]<<16 with n = nt*16 + (lane&15),
// k = ks*32 + (lane>>4)*16 + 2*i.  One fragment = 32 contiguous bytes/lane.
__global__ void k_repack_w(const float* __restrict__ W, unsigned int* __restrict__ Wp, int K) {
    int tid = blockIdx.x * blockDim.x + threadIdx.x;
    int total = K * 128;                 // K*256/2 packed uints
    if (tid >= total) return;
    int i    = tid & 7;
    int lane = (tid >> 3) & 31;
    int rest = tid >> 8;
    int nks  = K >> 5;
    int ks   = rest % nks;
    int nt   = rest / nks;
    int n = nt * 16 + (lane & 15);
    int k = ks * 32 + (lane >> 4) * 16 + 2 * i;
    unsigned int lo = f2bf(W[k * HID + n]);
    unsigned int h2 = f2bf(W[(k + 1) * HID + n]);
    Wp[tid] = lo | (h2 << 16);
}

// ---------------------- WMMA node GEMM: C[M,256] = A[M,256] @ W[256,256] ---
// 4 waves / block, block covers 64 rows x 256 cols. A tile staged in LDS as
// bf16 bits; B fragments (pre-swizzled) held in VGPRs across the 4 row-tiles.
__global__ __launch_bounds__(128) void k_gemm_node(
    const unsigned short* __restrict__ A,     // [M,256] bf16 bits
    const unsigned int* __restrict__ Wp,      // repacked [256,256] weights
    const float* __restrict__ bias,           // [256] or nullptr
    float* __restrict__ C,                    // [M,256] f32
    int M, int accumulate) {
    __shared__ unsigned short As[64 * 264];   // row stride 264 (pad, 16B-aligned)
    const int tid  = threadIdx.x;
    const int lane = tid & 31;
    const int wave = tid >> 5;
    const int row0 = blockIdx.x * 64;
    const bool full = (row0 + 64 <= M);       // uniform: all but the tail block

    // stage A (clamped rows for the ragged last block; stores skipped there)
    const unsigned int* Au = (const unsigned int*)A;
    unsigned int* Lu = (unsigned int*)As;
    for (int idx = tid; idx < 64 * 128; idx += 128) {
        int r = idx >> 7, c = idx & 127;
        int gr = row0 + r; if (gr >= M) gr = M - 1;
        Lu[r * 132 + c] = Au[(size_t)gr * 128 + c];
    }
    __syncthreads();

    const int mlo = lane & 15;
    const int hi  = lane >> 4;          // lane half

    for (int tsel = 0; tsel < 4; ++tsel) {
        const int nt = wave + tsel * 4;
        const int n  = nt * 16 + mlo;
        const float badd = bias ? bias[n] : 0.0f;
        // B fragments for all 8 K-steps: 2x b128 each, fully coalesced
        BFrag bf[8];
        #pragma unroll
        for (int ks = 0; ks < 8; ++ks) {
            const uint4* bp = (const uint4*)(Wp + (((nt * 8 + ks) * 32 + lane) << 3));
            bf[ks].q[0] = bp[0];
            bf[ks].q[1] = bp[1];
        }
        for (int rt = 0; rt < 4; ++rt) {
            v8f acc = {};
            const unsigned short* Arow = As + (rt * 16 + mlo) * 264;
            #pragma unroll
            for (int ks = 0; ks < 8; ++ks) {
                BFrag af;
                const uint4* ap = (const uint4*)(Arow + ks * 32 + hi * 16);
                af.q[0] = ap[0];        // k = ks*32 + hi*16 .. +8
                af.q[1] = ap[1];        // k = ks*32 + hi*16 + 8 .. +8
                acc = __builtin_amdgcn_wmma_f32_16x16x32_bf16(
                    false, af.v, false, bf[ks].v, (short)0, acc, false, false);
            }
            // one 64-bit base per tile; 8 stores at immediate offsets j*HID
            const int rbase = row0 + rt * 16 + hi * 8;
            float* Cb = C + (size_t)rbase * HID + n;
            if (full) {
                if (accumulate) {
                    #pragma unroll
                    for (int j = 0; j < 8; ++j)
                        Cb[j * HID] = Cb[j * HID] + acc[j] + badd;
                } else {
                    #pragma unroll
                    for (int j = 0; j < 8; ++j)
                        Cb[j * HID] = acc[j] + badd;
                }
            } else {
                #pragma unroll
                for (int j = 0; j < 8; ++j) {
                    if (rbase + j < M) {
                        float v = acc[j] + badd;
                        if (accumulate) v += Cb[j * HID];
                        Cb[j * HID] = v;
                    }
                }
            }
        }
    }
}

// ----------------- fused edge MLP: sigmoid(relu([h_s|h_d]W1+b1)W2+b2) ------
// 16 edges per block, 4 waves. Gathered concat rows staged to LDS as bf16,
// K=512 WMMA chain, ReLU in registers, 8-lane shuffle reduction for the
// final 256->1 projection.
__global__ __launch_bounds__(128) void k_edge_mlp(
    const float* __restrict__ h,
    const int* __restrict__ src, const int* __restrict__ dst,
    const unsigned int* __restrict__ We1p,    // repacked [512,256]
    const float* __restrict__ be1,            // [256]
    const float* __restrict__ We2,            // [256]
    const float* __restrict__ be2,            // [1]
    float* __restrict__ preds, int E) {
    __shared__ unsigned short Ae[16 * 520];   // 16 edges x 512 (pad 8, 16B-aligned)
    __shared__ float Hid[16 * 256];
    const int tid  = threadIdx.x;
    const int lane = tid & 31;
    const int wave = tid >> 5;
    const int e0   = blockIdx.x * 16;

    for (int idx = tid; idx < 16 * 512; idx += 128) {
        int e = idx >> 9;
        int f = idx & 511;
        size_t row = (size_t)((f < 256) ? src[e0 + e] : dst[e0 + e]);
        Ae[e * 520 + f] = f2bf(h[row * HID + (f & 255)]);
    }
    __syncthreads();

    const int mlo = lane & 15;
    const int hi  = lane >> 4;
    for (int tsel = 0; tsel < 4; ++tsel) {
        const int nt = wave + tsel * 4;
        const int n  = nt * 16 + mlo;
        v8f acc = {};
        const unsigned short* Arow = Ae + mlo * 520;
        #pragma unroll
        for (int ks = 0; ks < 16; ++ks) {
            BFrag af, bfv;
            const uint4* ap = (const uint4*)(Arow + ks * 32 + hi * 16);
            af.q[0] = ap[0];
            af.q[1] = ap[1];
            const uint4* bp = (const uint4*)(We1p + (((nt * 16 + ks) * 32 + lane) << 3));
            bfv.q[0] = bp[0];
            bfv.q[1] = bp[1];
            acc = __builtin_amdgcn_wmma_f32_16x16x32_bf16(
                false, af.v, false, bfv.v, (short)0, acc, false, false);
        }
        const float b = be1[n];
        float* Hb = Hid + (hi * 8) * 256 + n;
        #pragma unroll
        for (int j = 0; j < 8; ++j) {
            float v = acc[j] + b;
            Hb[j * 256] = v > 0.0f ? v : 0.0f;
        }
    }
    __syncthreads();

    // 256 -> 1 projection: 8 lanes per edge, shuffle reduce
    {
        int e = tid >> 3, seg = tid & 7;
        float s = 0.0f;
        int f0 = seg * 32;
        #pragma unroll 8
        for (int f = 0; f < 32; ++f) s += Hid[e * 256 + f0 + f] * We2[f0 + f];
        s += __shfl_xor(s, 4, 32);
        s += __shfl_xor(s, 2, 32);
        s += __shfl_xor(s, 1, 32);
        if (seg == 0) preds[e0 + e] = 1.0f / (1.0f + expf(-(s + be2[0])));
    }
}

// ---------------------------------------------------------------------------
extern "C" void kernel_launch(void* const* d_in, const int* in_sizes, int n_in,
                              void* d_out, int out_size, void* d_ws, size_t ws_size,
                              hipStream_t stream) {
    const int N = N_NODES, E = N_EDGES;
    const size_t NF = (size_t)N * HID;

    const float*     x   = (const float*)d_in[0];
    const long long* ei  = (const long long*)d_in[1];
    const float* Wx_l = (const float*)d_in[2];
    const float* bx_l = (const float*)d_in[3];
    const float* Wx_r = (const float*)d_in[4];
    const float* Wh_l = (const float*)d_in[5];
    const float* bh_l = (const float*)d_in[6];
    const float* Wh_r = (const float*)d_in[7];
    const float* Wb1  = (const float*)d_in[8];
    const float* bb1  = (const float*)d_in[9];
    const float* Wb2  = (const float*)d_in[10];
    const float* bb2  = (const float*)d_in[11];
    const float* We1  = (const float*)d_in[12];
    const float* be1  = (const float*)d_in[13];
    const float* We2  = (const float*)d_in[14];
    const float* be2  = (const float*)d_in[15];

    float* out_preds = (float*)d_out;          // [E]
    float* out_h     = (float*)d_out + E;      // [N,256]

    // workspace carve-up
    char* base = (char*)d_ws;
    size_t off = 0;
    auto carve = [&](size_t bytes) -> char* {
        char* p = base + off;
        off += (bytes + 255) & ~(size_t)255;
        return p;
    };
    float* F_H   = (float*)carve(NF * 4);
    float* F_HN  = (float*)carve(NF * 4);
    float* F_R   = (float*)carve(NF * 4);
    float* F_Z   = (float*)carve(NF * 4);
    float* F_HT  = (float*)carve(NF * 4);
    float* F_T   = (float*)carve(NF * 4);
    float* F_AGG = (float*)carve(NF * 4);
    float* DEG   = (float*)carve((size_t)N * 4);
    int*   SRC32 = (int*)carve((size_t)E * 4);
    int*   DST32 = (int*)carve((size_t)E * 4);
    unsigned short* B_X    = (unsigned short*)carve(NF * 2);
    unsigned short* B_AGGX = (unsigned short*)carve(NF * 2);
    unsigned short* B_H    = (unsigned short*)carve(NF * 2);
    unsigned short* B_AGG  = (unsigned short*)carve(NF * 2);
    unsigned short* B_HN   = (unsigned short*)carve(NF * 2);
    unsigned short* B_RH   = (unsigned short*)carve(NF * 2);
    const size_t WSQ  = (size_t)HID * HID;           // 65536 elements
    const size_t WPQ  = WSQ / 2;                     // packed uints per 256x256
    unsigned int* WXl = (unsigned int*)carve(3 * WSQ * 2);
    unsigned int* WXr = (unsigned int*)carve(3 * WSQ * 2);
    unsigned int* WHl = (unsigned int*)carve(4 * WSQ * 2);
    unsigned int* WHr = (unsigned int*)carve(4 * WSQ * 2);
    unsigned int* WB1 = (unsigned int*)carve(WSQ * 2);
    unsigned int* WB2 = (unsigned int*)carve(WSQ * 2);
    unsigned int* WE1 = (unsigned int*)carve(2 * WSQ * 2);   // [512,256]

    const int TPB = 256;
    const dim3 gNF((unsigned)((NF + TPB - 1) / TPB));
    const dim3 gN((unsigned)((N + TPB - 1) / TPB));
    const dim3 gE((unsigned)((E + TPB - 1) / TPB));
    const dim3 gScat((unsigned)(((size_t)E * 64 + TPB - 1) / TPB));
    const dim3 gGemm((unsigned)((N + 63) / 64));
    const dim3 gEdge((unsigned)(E / 16));

    auto zero  = [&](float* p, size_t n) {
        k_zero<<<(unsigned)((n + TPB - 1) / TPB), TPB, 0, stream>>>(p, n);
    };
    auto tobf  = [&](const float* s, unsigned short* d, size_t n) {
        k_f2bf<<<(unsigned)((n + TPB - 1) / TPB), TPB, 0, stream>>>(s, d, n);
    };
    auto repack = [&](const float* W, unsigned int* Wp, int K) {
        int total = K * 128;
        k_repack_w<<<(total + TPB - 1) / TPB, TPB, 0, stream>>>(W, Wp, K);
    };
    auto gemm  = [&](const unsigned short* A, const unsigned int* Wp,
                     const float* bias, float* C, int acc) {
        k_gemm_node<<<gGemm, 128, 0, stream>>>(A, Wp, bias, C, N, acc);
    };
    auto aggregate = [&](const float* X, unsigned short* out_bf) {
        zero(F_AGG, NF);
        k_scatter<<<gScat, TPB, 0, stream>>>(X, SRC32, DST32, F_AGG, E);
        k_scale_bf16<<<gNF, TPB, 0, stream>>>(F_AGG, DEG, out_bf, NF);
    };

    // ---- one-time prep ----
    k_i64_to_i32<<<gE, TPB, 0, stream>>>(ei,     SRC32, E);
    k_i64_to_i32<<<gE, TPB, 0, stream>>>(ei + E, DST32, E);
    zero(F_H, NF);                                       // h = 0
    zero(DEG, (size_t)N);
    k_deg<<<gE, TPB, 0, stream>>>(DST32, DEG, E);
    k_invdeg<<<gN, TPB, 0, stream>>>(DEG, N);            // DEG := inv_deg
    tobf(x, B_X, NF);
    aggregate(x, B_AGGX);                                // mean_agg(x) (constant)
    for (int s = 0; s < 3; ++s) {
        repack(Wx_l + s * WSQ, WXl + s * WPQ, HID);
        repack(Wx_r + s * WSQ, WXr + s * WPQ, HID);
    }
    for (int s = 0; s < 4; ++s) {
        repack(Wh_l + s * WSQ, WHl + s * WPQ, HID);
        repack(Wh_r + s * WSQ, WHr + s * WPQ, HID);
    }
    repack(Wb1, WB1, HID);
    repack(Wb2, WB2, HID);
    repack(We1, WE1, 2 * HID);

    // ---- 2 GRU-SAGE layers ----
    for (int layer = 0; layer < 2; ++layer) {
        tobf(F_H, B_H, NF);
        // h_N = sage(h; Wh_l[0], bh_l[0], Wh_r[0])
        aggregate(F_H, B_AGG);
        gemm(B_AGG, WHl + 0 * WPQ, bh_l + 0 * HID, F_HN, 0);
        gemm(B_H,   WHr + 0 * WPQ, nullptr,        F_HN, 1);
        // h_N := h + tanh(h@Wb1 + bb1 + h_N@Wb2 + bb2)
        gemm(B_H, WB1, bb1, F_T, 0);
        tobf(F_HN, B_HN, NF);
        gemm(B_HN, WB2, bb2, F_T, 1);
        k_add_tanh<<<gNF, TPB, 0, stream>>>(F_HN, F_H, F_T, NF);
        tobf(F_HN, B_HN, NF);
        aggregate(F_HN, B_AGG);                          // mean_agg(h_N)
        // r = sigmoid(sage(x;0) + sage(h_N;1))
        gemm(B_AGGX, WXl + 0 * WPQ, bx_l + 0 * HID, F_R, 0);
        gemm(B_X,    WXr + 0 * WPQ, nullptr,        F_R, 1);
        gemm(B_AGG,  WHl + 1 * WPQ, bh_l + 1 * HID, F_R, 1);
        gemm(B_HN,   WHr + 1 * WPQ, nullptr,        F_R, 1);
        k_sigmoid<<<gNF, TPB, 0, stream>>>(F_R, NF);
        // z = sigmoid(sage(x;1) + sage(h_N;2))
        gemm(B_AGGX, WXl + 1 * WPQ, bx_l + 1 * HID, F_Z, 0);
        gemm(B_X,    WXr + 1 * WPQ, nullptr,        F_Z, 1);
        gemm(B_AGG,  WHl + 2 * WPQ, bh_l + 2 * HID, F_Z, 1);
        gemm(B_HN,   WHr + 2 * WPQ, nullptr,        F_Z, 1);
        k_sigmoid<<<gNF, TPB, 0, stream>>>(F_Z, NF);
        // h_tilde = tanh(sage(x;2) + sage(r*h_N;3))
        k_mul<<<gNF, TPB, 0, stream>>>(F_T, F_R, F_HN, NF);
        tobf(F_T, B_RH, NF);
        aggregate(F_T, B_AGG);                           // mean_agg(r*h_N)
        gemm(B_AGGX, WXl + 2 * WPQ, bx_l + 2 * HID, F_HT, 0);
        gemm(B_X,    WXr + 2 * WPQ, nullptr,        F_HT, 1);
        gemm(B_AGG,  WHl + 3 * WPQ, bh_l + 3 * HID, F_HT, 1);
        gemm(B_RH,   WHr + 3 * WPQ, nullptr,        F_HT, 1);
        k_tanh<<<gNF, TPB, 0, stream>>>(F_HT, NF);
        // h = (1-z)*h_N + z*h_tilde
        k_gate<<<gNF, TPB, 0, stream>>>(F_H, F_Z, F_HN, F_HT, NF);
    }

    // ---- outputs ----
    k_copy<<<gNF, TPB, 0, stream>>>(out_h, F_H, NF);
    k_edge_mlp<<<gEdge, 128, 0, stream>>>(F_H, SRC32, DST32, WE1, be1, We2, be2,
                                          out_preds, E);
}
